// FunctionBoundaryModel_30219389895250
// MI455X (gfx1250) — matrix operand order repaired and checked by reading, not compiled, not run on hardware
//
#include <hip/hip_runtime.h>
#include <hip/hip_bf16.h>
#include <stdint.h>

#define B_  4
#define L_  4096
#define DM  512
#define DI  1024
#define NS  16
#define DR  32
#define KC  4
#define NL  4
#define M_  (B_ * L_)
#define NC  64            /* scan chunks (must be 64: shifts below assume it) */
#define LC  (L_ / NC)

typedef __attribute__((ext_vector_type(16))) __bf16 v16bf;
typedef __attribute__((ext_vector_type(8)))  float  v8f;

static __device__ __forceinline__ __bf16 f2bf(float f) {
  uint32_t u = __builtin_bit_cast(uint32_t, f);
  uint32_t r = u + 0x7FFFu + ((u >> 16) & 1u);   // round-to-nearest-even
  uint16_t h = (uint16_t)(r >> 16);
  return __builtin_bit_cast(__bf16, h);
}
static __device__ __forceinline__ float bf2f(__bf16 b) {
  uint32_t u = ((uint32_t)__builtin_bit_cast(uint16_t, b)) << 16;
  return __builtin_bit_cast(float, u);
}
static __device__ __forceinline__ float fast_sigmoid(float x) {
  return 1.0f / (1.0f + exp2f(-1.44269504f * x));
}

// ---------------------------------------------------------------- converts
__global__ void k_cvt_bf16(const float* __restrict__ src, __bf16* __restrict__ dst, int n) {
  int i = blockIdx.x * 256 + threadIdx.x;
  if (i < n) dst[i] = f2bf(src[i]);
}

// gather embeddings -> bf16 A-operand (M_ x 256)
__global__ void k_embed(const int* __restrict__ tok, const float* __restrict__ emb,
                        __bf16* __restrict__ out) {
  int i = blockIdx.x * 256 + threadIdx.x;   // over M_*256
  int m = i >> 8, e = i & 255;
  out[i] = f2bf(emb[tok[m] * 256 + e]);
}

// ---------------------------------------------------------------- layernorm
// one wave32 per 512-wide row; writes bf16 for the following WMMA GEMM
__global__ void k_layernorm(const float* __restrict__ x, const float* __restrict__ w,
                            const float* __restrict__ b, __bf16* __restrict__ out) {
  int wid  = blockIdx.x * 8 + (threadIdx.x >> 5);
  int lane = threadIdx.x & 31;
  const float* row = x + (size_t)wid * DM;
  float v[16];
  float s = 0.0f, ss = 0.0f;
#pragma unroll
  for (int i = 0; i < 16; i++) {
    float t = row[lane + i * 32];
    v[i] = t; s += t; ss += t * t;
  }
#pragma unroll
  for (int m = 16; m >= 1; m >>= 1) {
    s  += __shfl_xor(s,  m, 32);
    ss += __shfl_xor(ss, m, 32);
  }
  float mu   = s * (1.0f / (float)DM);
  float var  = ss * (1.0f / (float)DM) - mu * mu;
  float rstd = rsqrtf(var + 1e-5f);
  __bf16* orow = out + (size_t)wid * DM;
#pragma unroll
  for (int i = 0; i < 16; i++) {
    int c = lane + i * 32;
    orow[c] = f2bf((v[i] - mu) * rstd * w[c] + b[c]);
  }
}

// ---------------------------------------------------------------- WMMA GEMM
// C[M_,N] = epilogue(A[M_,K](bf16) @ W[N,K]^T(bf16) + bias) (+ Cadd)
// one wave per 32x32 C block (2x2 of 16x16 WMMA tiles): 4 WMMAs share
// 2 A-fragments + 2 B-fragments -> 2x the FLOP/byte of a 1-tile wave.
// op: 0 none, 1 softplus, 2 relu.  Cb!=null -> bf16 output, else fp32 to Cf.
__global__ void k_gemm(const __bf16* __restrict__ A, const __bf16* __restrict__ W,
                       const float* __restrict__ bias, const float* __restrict__ Cadd,
                       float* __restrict__ Cf, __bf16* __restrict__ Cb,
                       int Ntiles, int N, int K, int op) {
  int wid  = blockIdx.x * 8 + (threadIdx.x >> 5);
  int lane = threadIdx.x & 31;
  int rt   = wid / Ntiles;
  int ct   = wid - rt * Ntiles;
  int row0 = rt << 5, col0 = ct << 5;
  int half = lane >> 4, lm = lane & 15;
  int n0 = col0 + lm;                 // B columns held by this lane
  int n1 = col0 + 16 + lm;
  bool n0ok = (n0 < N), n1ok = (n1 < N);

  v8f acc00 = {}, acc01 = {}, acc10 = {}, acc11 = {};
  const uint32_t* A0 = (const uint32_t*)(A + (size_t)(row0 + lm) * K);
  const uint32_t* A1 = (const uint32_t*)(A + (size_t)(row0 + 16 + lm) * K);
  const __bf16*   W0 = W + (size_t)(n0ok ? n0 : 0) * K + half * 16;
  const __bf16*   W1 = W + (size_t)(n1ok ? n1 : 0) * K + half * 16;

  for (int k0 = 0; k0 < K; k0 += 32) {
    // A fragment: 16x32 bf16, lane holds row (lane&15); 16 contiguous
    // K values starting at k0 + half*16 (matches B fragment order)
    union { v16bf v; uint32_t u[8]; } a0, a1;
    const uint32_t* p0 = A0 + (k0 >> 1) + half * 8;
    const uint32_t* p1 = A1 + (k0 >> 1) + half * 8;
#pragma unroll
    for (int j = 0; j < 4; j++) {
      a0.u[j] = p0[j]; a0.u[4 + j] = p0[4 + j];
      a1.u[j] = p1[j]; a1.u[4 + j] = p1[4 + j];
    }
    // B fragment: 32x16 bf16, lane holds column (lane&15); 16 contiguous
    // K values starting at k0 + half*16 (row-major weight (N,K))
    v16bf b0 = *(const v16bf*)(W0 + k0);
    v16bf b1 = *(const v16bf*)(W1 + k0);
    if (k0 + 32 < K) {                       // prefetch next K-step
      __builtin_prefetch((const void*)(p0 + 16), 0, 0);
      __builtin_prefetch((const void*)(W0 + k0 + 32), 0, 0);
    }
    acc00 = __builtin_amdgcn_wmma_f32_16x16x32_bf16(false, a0.v, false, b0,
                                                    (short)0, acc00, false, false);
    acc01 = __builtin_amdgcn_wmma_f32_16x16x32_bf16(false, a0.v, false, b1,
                                                    (short)0, acc01, false, false);
    acc10 = __builtin_amdgcn_wmma_f32_16x16x32_bf16(false, a1.v, false, b0,
                                                    (short)0, acc10, false, false);
    acc11 = __builtin_amdgcn_wmma_f32_16x16x32_bf16(false, a1.v, false, b1,
                                                    (short)0, acc11, false, false);
  }

  auto store_tile = [&](v8f acc, int r0, int n, bool nok) {
    if (!nok) return;
    float bv = bias ? bias[n] : 0.0f;
#pragma unroll
    for (int i = 0; i < 8; i++) {
      float vv = acc[i] + bv;
      if (op == 1)      vv = fmaxf(vv, 0.0f) + log1pf(expf(-fabsf(vv)));  // softplus
      else if (op == 2) vv = fmaxf(vv, 0.0f);                             // relu
      size_t offo = (size_t)(r0 + i + half * 8) * N + n;  // C VGPR i -> row i + 8*half
      if (Cadd) vv += Cadd[offo];
      if (Cb) Cb[offo] = f2bf(vv);
      else    Cf[offo] = vv;
    }
  };
  store_tile(acc00, row0,      n0, n0ok);
  store_tile(acc01, row0,      n1, n1ok);
  store_tile(acc10, row0 + 16, n0, n0ok);
  store_tile(acc11, row0 + 16, n1, n1ok);
}

// ---------------------------------------------------------------- conv+silu
// causal depthwise conv1d (KC=4) over the x-half of xz, then SiLU -> bf16
__global__ void k_conv_silu(const __bf16* __restrict__ xz, const float* __restrict__ cw,
                            const float* __restrict__ cb, __bf16* __restrict__ xc) {
  int i = blockIdx.x * 256 + threadIdx.x;   // over M_*DI
  int m = i >> 10;
  int d = i & (DI - 1);
  int l = m & (L_ - 1);
  float acc = cb[d];
#pragma unroll
  for (int t = 0; t < KC; t++) {
    int lt = l + t - (KC - 1);
    if (lt >= 0)
      acc += cw[d * KC + t] * bf2f(xz[(size_t)(m + t - (KC - 1)) * (2 * DI) + d]);
  }
  acc = acc * fast_sigmoid(acc);
  xc[i] = f2bf(acc);
}

// extract dt_low (first DR cols of xdbc) -> bf16 for the dt GEMM
__global__ void k_dtlow(const float* __restrict__ xdbc, __bf16* __restrict__ dtl) {
  int i = blockIdx.x * 256 + threadIdx.x;   // over M_*DR
  int m = i >> 5;
  int r = i & (DR - 1);
  dtl[i] = f2bf(xdbc[(size_t)m * 64 + r]);
}

// ---------------------------------------------------------------- SSM scan
// Linear recurrence h_t = exp(dt_t*A)*h_{t-1} + (dt_t*x_t)*B_t, chunked 3-pass.
// thread id layout: tid = ((b*NC + c) * DI) + d  (d fastest => coalesced dt/x)

__global__ void k_scan1(const float* __restrict__ dt, const __bf16* __restrict__ xc,
                        const float* __restrict__ xdbc, const float* __restrict__ A_log,
                        float* __restrict__ hend, float* __restrict__ Ssum) {
  int tid = blockIdx.x * 256 + threadIdx.x;   // B_*NC*DI
  int d  = tid & (DI - 1);
  int bc = tid >> 10;
  int c  = bc & (NC - 1);
  int b  = bc >> 6;                            // NC == 64
  float a2[NS];
#pragma unroll
  for (int nn = 0; nn < NS; nn++) a2[nn] = -expf(A_log[d * NS + nn]) * 1.44269504f;
  float h[NS];
#pragma unroll
  for (int nn = 0; nn < NS; nn++) h[nn] = 0.0f;
  float S = 0.0f;
  int base_l = c * LC;
  for (int t = 0; t < LC; t++) {
    size_t idx = (size_t)(b * L_ + base_l + t);
    float dtv = dt[idx * DI + d];
    float xv  = bf2f(xc[idx * DI + d]);
    S += dtv;
    float dtx = dtv * xv;
    const float* bp = xdbc + idx * 64 + DR;   // B_t[0..15]
#pragma unroll
    for (int nn = 0; nn < NS; nn++) {
      float dA = exp2f(dtv * a2[nn]);
      h[nn] = dA * h[nn] + dtx * bp[nn];
    }
  }
  size_t hb = ((size_t)((b * DI + d) * NC + c)) * NS;
#pragma unroll
  for (int nn = 0; nn < NS; nn++) hend[hb + nn] = h[nn];
  Ssum[(b * DI + d) * NC + c] = S;
}

// inter-chunk recurrence: chunk-product of exp(dt*A) == exp2(a2 * sum(dt))
__global__ void k_scan2(const float* __restrict__ hend, const float* __restrict__ Ssum,
                        const float* __restrict__ A_log, float* __restrict__ Hinit) {
  int tid = blockIdx.x * 256 + threadIdx.x;   // B_*DI  (== b*DI + d)
  int d = tid & (DI - 1);
  float a2[NS];
#pragma unroll
  for (int nn = 0; nn < NS; nn++) a2[nn] = -expf(A_log[d * NS + nn]) * 1.44269504f;
  float H[NS];
#pragma unroll
  for (int nn = 0; nn < NS; nn++) H[nn] = 0.0f;
  size_t base = (size_t)tid * NC;
  for (int c = 0; c < NC; c++) {
    size_t hb = (base + c) * NS;
#pragma unroll
    for (int nn = 0; nn < NS; nn++) Hinit[hb + nn] = H[nn];
    float S = Ssum[base + c];
#pragma unroll
    for (int nn = 0; nn < NS; nn++) H[nn] = exp2f(S * a2[nn]) * H[nn] + hend[hb + nn];
  }
}

// replay with corrected initial state; fuse y = h.C + x*Dp, gate with silu(z),
// and emit bf16 directly for the out-projection GEMM
__global__ void k_scan3(const float* __restrict__ dt, const __bf16* __restrict__ xc,
                        const float* __restrict__ xdbc, const __bf16* __restrict__ xz,
                        const float* __restrict__ A_log, const float* __restrict__ Dp,
                        const float* __restrict__ Hinit, __bf16* __restrict__ g) {
  int tid = blockIdx.x * 256 + threadIdx.x;   // B_*NC*DI
  int d  = tid & (DI - 1);
  int bc = tid >> 10;
  int c  = bc & (NC - 1);
  int b  = bc >> 6;
  float a2[NS];
#pragma unroll
  for (int nn = 0; nn < NS; nn++) a2[nn] = -expf(A_log[d * NS + nn]) * 1.44269504f;
  float h[NS];
  size_t hb = ((size_t)((b * DI + d) * NC + c)) * NS;
#pragma unroll
  for (int nn = 0; nn < NS; nn++) h[nn] = Hinit[hb + nn];
  float dp = Dp[d];
  int base_l = c * LC;
  for (int t = 0; t < LC; t++) {
    size_t idx = (size_t)(b * L_ + base_l + t);
    float dtv = dt[idx * DI + d];
    float xv  = bf2f(xc[idx * DI + d]);
    float dtx = dtv * xv;
    const float* bp = xdbc + idx * 64 + DR;   // B_t
    const float* cp = bp + NS;                // C_t
    float y = 0.0f;
#pragma unroll
    for (int nn = 0; nn < NS; nn++) {
      float dA = exp2f(dtv * a2[nn]);
      h[nn] = dA * h[nn] + dtx * bp[nn];
      y += h[nn] * cp[nn];
    }
    y += xv * dp;
    float zv = bf2f(xz[idx * (2 * DI) + DI + d]);
    g[idx * DI + d] = f2bf(y * zv * fast_sigmoid(zv));
  }
}

// ---------------------------------------------------------------- host
extern "C" void kernel_launch(void* const* d_in, const int* in_sizes, int n_in,
                              void* d_out, int out_size, void* d_ws, size_t ws_size,
                              hipStream_t stream) {
  (void)in_sizes; (void)n_in; (void)out_size; (void)ws_size;
  const int*   tokens    = (const int*)d_in[0];
  const float* emb       = (const float*)d_in[1];
  const float* inp_w     = (const float*)d_in[2];
  const float* inp_b     = (const float*)d_in[3];
  const float* norm_w    = (const float*)d_in[4];
  const float* norm_b    = (const float*)d_in[5];
  const float* in_proj_w = (const float*)d_in[6];
  const float* conv_w    = (const float*)d_in[7];
  const float* conv_b    = (const float*)d_in[8];
  const float* x_proj_w  = (const float*)d_in[9];
  const float* dt_w      = (const float*)d_in[10];
  const float* dt_b      = (const float*)d_in[11];
  const float* A_log     = (const float*)d_in[12];
  const float* Dp        = (const float*)d_in[13];
  const float* out_w     = (const float*)d_in[14];
  const float* fnorm_w   = (const float*)d_in[15];
  const float* fnorm_b   = (const float*)d_in[16];
  const float* h1_w      = (const float*)d_in[17];
  const float* h1_b      = (const float*)d_in[18];
  const float* h2_w      = (const float*)d_in[19];
  const float* h2_b      = (const float*)d_in[20];
  float* outp = (float*)d_out;

  char*  ws  = (char*)d_ws;
  size_t off = 0;
  auto alloc = [&](size_t bytes) -> void* {
    void* p = ws + off;
    off = (off + bytes + 255) & ~(size_t)255;
    return p;
  };

  __bf16* wi_bf   = (__bf16*)alloc((size_t)NL * 2 * DI * DM * 2);
  __bf16* wx_bf   = (__bf16*)alloc((size_t)NL * 64 * DI * 2);
  __bf16* wdt_bf  = (__bf16*)alloc((size_t)NL * DI * DR * 2);
  __bf16* wo_bf   = (__bf16*)alloc((size_t)NL * DM * DI * 2);
  __bf16* winp_bf = (__bf16*)alloc((size_t)DM * 256 * 2);
  __bf16* wh1_bf  = (__bf16*)alloc((size_t)256 * DM * 2);
  __bf16* wh2_bf  = (__bf16*)alloc((size_t)5 * 256 * 2);
  __bf16* embg    = (__bf16*)alloc((size_t)M_ * 256 * 2);
  __bf16* hbf     = (__bf16*)alloc((size_t)M_ * DM * 2);
  __bf16* xz      = (__bf16*)alloc((size_t)M_ * 2 * DI * 2);
  __bf16* xcbf    = (__bf16*)alloc((size_t)M_ * DI * 2);
  float*  xdbc    = (float*)alloc((size_t)M_ * 64 * 4);
  __bf16* dtl     = (__bf16*)alloc((size_t)M_ * DR * 2);
  float*  dtf     = (float*)alloc((size_t)M_ * DI * 4);
  __bf16* gbf     = (__bf16*)alloc((size_t)M_ * DI * 2);
  __bf16* t1bf    = (__bf16*)alloc((size_t)M_ * 256 * 2);
  float*  xres    = (float*)alloc((size_t)M_ * DM * 4);
  float*  Ssum    = (float*)alloc((size_t)B_ * DI * NC * 4);
  float*  hendp   = (float*)alloc((size_t)B_ * DI * NC * NS * 4);
  float*  Hinit   = (float*)alloc((size_t)B_ * DI * NC * NS * 4);

  auto cvt = [&](const float* s, __bf16* dst, size_t n) {
    k_cvt_bf16<<<(unsigned)((n + 255) / 256), 256, 0, stream>>>(s, dst, (int)n);
  };
  cvt(in_proj_w, wi_bf,  (size_t)NL * 2 * DI * DM);
  cvt(x_proj_w,  wx_bf,  (size_t)NL * 64 * DI);
  cvt(dt_w,      wdt_bf, (size_t)NL * DI * DR);
  cvt(out_w,     wo_bf,  (size_t)NL * DM * DI);
  cvt(inp_w,     winp_bf,(size_t)DM * 256);
  cvt(h1_w,      wh1_bf, (size_t)256 * DM);
  cvt(h2_w,      wh2_bf, (size_t)5 * 256);

  k_embed<<<(M_ * 256) / 256, 256, 0, stream>>>(tokens, emb, embg);

  auto gemm = [&](const __bf16* A, const __bf16* W, const float* bias, const float* Cadd,
                  float* Cf, __bf16* Cb, int N, int K, int op) {
    int Ntiles = (N + 31) / 32;
    int tiles  = (M_ / 32) * Ntiles;    // M_/32 = 512 -> multiple of 8
    k_gemm<<<tiles / 8, 256, 0, stream>>>(A, W, bias, Cadd, Cf, Cb, Ntiles, N, K, op);
  };

  // x = emb[tokens] @ inp_w.T + inp_b  -> residual stream (fp32)
  gemm(embg, winp_bf, inp_b, nullptr, xres, nullptr, DM, 256, 0);

  for (int i = 0; i < NL; i++) {
    const float* Ai = A_log + (size_t)i * DI * NS;
    k_layernorm<<<M_ / 8, 256, 0, stream>>>(xres, norm_w + i * DM, norm_b + i * DM, hbf);
    gemm(hbf, wi_bf + (size_t)i * 2 * DI * DM, nullptr, nullptr, nullptr, xz, 2 * DI, DM, 0);
    k_conv_silu<<<(M_ * DI) / 256, 256, 0, stream>>>(xz, conv_w + i * DI * KC,
                                                     conv_b + i * DI, xcbf);
    gemm(xcbf, wx_bf + (size_t)i * 64 * DI, nullptr, nullptr, xdbc, nullptr, 64, DI, 0);
    k_dtlow<<<(M_ * DR) / 256, 256, 0, stream>>>(xdbc, dtl);
    gemm(dtl, wdt_bf + (size_t)i * DI * DR, dt_b + i * DI, nullptr, dtf, nullptr, DI, DR, 1);
    k_scan1<<<(B_ * NC * DI) / 256, 256, 0, stream>>>(dtf, xcbf, xdbc, Ai, hendp, Ssum);
    k_scan2<<<(B_ * DI) / 256, 256, 0, stream>>>(hendp, Ssum, Ai, Hinit);
    k_scan3<<<(B_ * NC * DI) / 256, 256, 0, stream>>>(dtf, xcbf, xdbc, xz, Ai,
                                                      Dp + i * DI, Hinit, gbf);
    gemm(gbf, wo_bf + (size_t)i * DM * DI, nullptr, xres, xres, nullptr, DM, DI, 0);
  }

  k_layernorm<<<M_ / 8, 256, 0, stream>>>(xres, fnorm_w, fnorm_b, hbf);
  gemm(hbf, wh1_bf, h1_b, nullptr, nullptr, t1bf, 256, DM, 2);   // relu -> bf16
  gemm(t1bf, wh2_bf, h2_b, nullptr, outp, nullptr, 5, 256, 0);   // logits fp32
}
